// GATModel_57363583205983
// MI455X (gfx1250) — matrix operand order repaired and checked by reading, not compiled, not run on hardware
//
#include <hip/hip_runtime.h>
#include <hip/hip_bf16.h>
#include <math.h>

typedef __attribute__((ext_vector_type(16))) _Float16 v16h;
typedef __attribute__((ext_vector_type(8)))  _Float16 v8h;
typedef __attribute__((ext_vector_type(8)))  float    v8f;

// ---------------------------------------------------------------------------
// CSR build (dst-sorted edge list, self loops appended)
// ---------------------------------------------------------------------------
__global__ void count_edges_kernel(const int* __restrict__ ei, int E, int n_nodes,
                                   int* __restrict__ deg) {
  int e = blockIdx.x * blockDim.x + threadIdx.x;
  int total = E + n_nodes;
  if (e >= total) return;
  int dst = (e < E) ? ei[E + e] : (e - E);   // ei layout: [0..E)=src, [E..2E)=dst
  atomicAdd(&deg[dst], 1);
}

// Single-block chunked exclusive scan: deg[n] -> row_ptr[n+1], cursor[n].
// deg may alias cursor (each index read before written, same iteration).
__global__ void scan_kernel(const int* deg, int* row_ptr, int* cursor, int n) {
  __shared__ int sdata[1024];
  __shared__ int carry_s;
  int tid = threadIdx.x;
  if (tid == 0) carry_s = 0;
  __syncthreads();
  for (int base = 0; base < n; base += 1024) {
    int i = base + tid;
    int v = (i < n) ? deg[i] : 0;
    sdata[tid] = v;
    __syncthreads();
    #pragma unroll
    for (int off = 1; off < 1024; off <<= 1) {
      int t = (tid >= off) ? sdata[tid - off] : 0;
      __syncthreads();
      sdata[tid] += t;
      __syncthreads();
    }
    int incl = sdata[tid];
    int carry = carry_s;
    if (i < n) {
      int excl = carry + incl - v;
      row_ptr[i] = excl;
      cursor[i]  = excl;
    }
    __syncthreads();
    if (tid == 1023) carry_s = carry + incl;
    __syncthreads();
  }
  if (tid == 0) row_ptr[n] = carry_s;
}

__global__ void scatter_edges_kernel(const int* __restrict__ ei, int E, int n_nodes,
                                     int* __restrict__ cursor, int* __restrict__ col_src) {
  int e = blockIdx.x * blockDim.x + threadIdx.x;
  int total = E + n_nodes;
  if (e >= total) return;
  int src, dst;
  if (e < E) { src = ei[e]; dst = ei[E + e]; }
  else       { src = e - E; dst = e - E; }
  int pos = atomicAdd(&cursor[dst], 1);
  col_src[pos] = src;
}

// ---------------------------------------------------------------------------
// Precision prep: f32 -> f16 (row-major, pairwise packed stores)
// ---------------------------------------------------------------------------
__global__ void f32_to_f16_kernel(const float* __restrict__ in,
                                  _Float16* __restrict__ out, long long npairs) {
  long long i = (long long)blockIdx.x * blockDim.x + threadIdx.x;
  if (i >= npairs) return;
  float2 v = ((const float2*)in)[i];
  union { _Float16 h[2]; unsigned u; } u;
  u.h[0] = (_Float16)v.x; u.h[1] = (_Float16)v.y;
  ((unsigned*)out)[i] = u.u;
}

// Pack B [K,64] f32 into WMMA B-operand lane layout, f16:
// slot idx = (kt*4 + nt)*32 + lane ; halves j=0..15 :
//   B[kt*32 + (lane>=16?16:0) + j][nt*16 + (lane&15)]
__global__ void pack_b_kernel(const float* __restrict__ B,
                              _Float16* __restrict__ out, int K) {
  int idx = blockIdx.x * blockDim.x + threadIdx.x;
  int nslots = (K >> 5) * 4 * 32;
  if (idx >= nslots) return;
  int lane = idx & 31;
  int nt   = (idx >> 5) & 3;
  int kt   = idx >> 7;
  int n  = nt * 16 + (lane & 15);
  int kb = kt * 32 + ((lane >= 16) ? 16 : 0);
  _Float16* o = out + (size_t)idx * 16;
  #pragma unroll
  for (int j = 0; j < 16; ++j) o[j] = (_Float16)B[(size_t)(kb + j) * 64 + n];
}

// ---------------------------------------------------------------------------
// WMMA GEMM: C[M,64] = A[M,K](f16 row-major) * Bp(pre-packed f16), f32 out.
// One wave -> 16 rows x 64 cols (four 16x16 f32 accumulators), K%32 == 0.
// Inner loop: 2x b128 A loads, 4x 32B B loads, 4x v_wmma.
// ---------------------------------------------------------------------------
__global__ __launch_bounds__(128) void gemm_wmma_f16(
    const _Float16* __restrict__ A, const _Float16* __restrict__ Bp,
    float* __restrict__ C, int M, int K) {
  int wave = blockIdx.x * (blockDim.x >> 5) + (threadIdx.x >> 5);
  int lane = threadIdx.x & 31;
  int row0 = wave * 16;
  if (row0 >= M) return;

  bool hi = lane >= 16;
  int arow_i = row0 + (lane & 15);
  if (arow_i >= M) arow_i = M - 1;           // clamp (stores are guarded)
  const _Float16* arow = A + (size_t)arow_i * K;
  const v16h* bp = (const v16h*)Bp;

  v8f acc[4] = {v8f{}, v8f{}, v8f{}, v8f{}};

  int ktiles = K >> 5;
  for (int kt = 0; kt < ktiles; ++kt) {
    // A tile 16x32 f16: low lanes K {0..15}, high lanes K {16..31}
    int kb = kt * 32 + (hi ? 16 : 0);
    v8h alo = *(const v8h*)(arow + kb);
    v8h ahi = *(const v8h*)(arow + kb + 8);
    v16h a = __builtin_shufflevector(alo, ahi,
                                     0, 1, 2, 3, 4, 5, 6, 7,
                                     8, 9, 10, 11, 12, 13, 14, 15);
    const v16h* bt = bp + (size_t)kt * 128 + lane;
    #pragma unroll
    for (int t = 0; t < 4; ++t) {
      acc[t] = __builtin_amdgcn_wmma_f32_16x16x32_f16(
          false, a, false, bt[t * 32], (short)0, acc[t], false, false);
    }
  }

  int rbase = row0 + (hi ? 8 : 0);
  #pragma unroll
  for (int t = 0; t < 4; ++t) {
    int n = t * 16 + (lane & 15);
    #pragma unroll
    for (int v = 0; v < 8; ++v) {
      int r = rbase + v;
      if (r < M) C[(size_t)r * 64 + n] = acc[t][v];
    }
  }
}

// ---------------------------------------------------------------------------
// GATv2 aggregation, one wave per dst node, online softmax over CSR edges.
// 64 channels = 2 per lane; head h = lane>>3; logit reduced over 8-lane group.
// Output: f32 (outf) or f16 (outh) for feeding the next GEMM directly.
// ---------------------------------------------------------------------------
__global__ __launch_bounds__(128) void gat_aggregate(
    const float* __restrict__ xl, const float* __restrict__ xr,
    const int* __restrict__ row_ptr, const int* __restrict__ col_src,
    const float* __restrict__ att, const float* __restrict__ bias,
    float* __restrict__ outf, _Float16* __restrict__ outh, int n, int do_relu) {
  int node = blockIdx.x * (blockDim.x >> 5) + (threadIdx.x >> 5);
  if (node >= n) return;
  int lane = threadIdx.x & 31;
  int c0 = 2 * lane, c1 = c0 + 1;
  int h = lane >> 3;
  int cin = 2 * (lane & 7);
  float a0 = att[h * 16 + cin];
  float a1 = att[h * 16 + cin + 1];
  float2 xrv = *(const float2*)(xr + (size_t)node * 64 + c0);

  int beg = row_ptr[node], end = row_ptr[node + 1];
  float m = -INFINITY, s = 0.f, acc0 = 0.f, acc1 = 0.f;
  for (int idx = beg; idx < end; ++idx) {
    int srcn = col_src[idx];
    float2 xlv = *(const float2*)(xl + (size_t)srcn * 64 + c0);
    float e0 = xlv.x + xrv.x; e0 = e0 > 0.f ? e0 : 0.2f * e0;   // LeakyReLU 0.2
    float e1 = xlv.y + xrv.y; e1 = e1 > 0.f ? e1 : 0.2f * e1;
    float p = a0 * e0 + a1 * e1;
    p += __shfl_xor(p, 1);
    p += __shfl_xor(p, 2);
    p += __shfl_xor(p, 4);          // all 8 lanes of this head hold the logit
    float nm    = fmaxf(m, p);
    float scale = __expf(m - nm);   // first iter: exp(-inf) = 0
    float w     = __expf(p - nm);
    s    = s * scale + w;
    acc0 = acc0 * scale + w * xlv.x;
    acc1 = acc1 * scale + w * xlv.y;
    m = nm;
  }
  float inv = 1.0f / s;             // self loop guarantees s > 0
  float o0 = acc0 * inv + bias[c0];
  float o1 = acc1 * inv + bias[c1];
  if (do_relu) { o0 = fmaxf(o0, 0.f); o1 = fmaxf(o1, 0.f); }
  if (outh) {
    union { _Float16 hh[2]; unsigned u; } u;
    u.hh[0] = (_Float16)o0; u.hh[1] = (_Float16)o1;
    ((unsigned*)(outh + (size_t)node * 64))[lane] = u.u;
  } else {
    float2 st; st.x = o0; st.y = o1;
    *(float2*)(outf + (size_t)node * 64 + c0) = st;
  }
}

// ---------------------------------------------------------------------------
extern "C" void kernel_launch(void* const* d_in, const int* in_sizes, int n_in,
                              void* d_out, int out_size, void* d_ws, size_t ws_size,
                              hipStream_t stream) {
  const float* x    = (const float*)d_in[0];
  const int*   ei   = (const int*)d_in[1];
  const float* w_l1 = (const float*)d_in[2];
  const float* w_r1 = (const float*)d_in[3];
  const float* att1 = (const float*)d_in[4];
  const float* b1   = (const float*)d_in[5];
  const float* w_l2 = (const float*)d_in[6];
  const float* w_r2 = (const float*)d_in[7];
  const float* att2 = (const float*)d_in[8];
  const float* b2   = (const float*)d_in[9];
  float* out = (float*)d_out;

  const int F_IN = in_sizes[2] / 64;        // 128
  const int N    = in_sizes[0] / F_IN;      // 100000
  const int E    = in_sizes[1] / 2;         // 1600000
  const int Etot = E + N;                   // with self loops

  // Workspace carve-up (256B-aligned bump allocator)
  uintptr_t base = (uintptr_t)d_ws;
  auto alloc = [&](size_t bytes) -> void* {
    uintptr_t p = (base + 255) & ~(uintptr_t)255;
    base = p + bytes;
    return (void*)p;
  };
  float*    xl      = (float*)alloc((size_t)N * 64 * sizeof(float));
  float*    xr      = (float*)alloc((size_t)N * 64 * sizeof(float));
  _Float16* xh      = (_Float16*)alloc((size_t)N * F_IN * sizeof(_Float16));
  _Float16* h16     = (_Float16*)alloc((size_t)N * 64 * sizeof(_Float16));
  int*      row_ptr = (int*)alloc((size_t)(N + 1) * sizeof(int));
  int*      cursor  = (int*)alloc((size_t)N * sizeof(int));       // also deg
  int*      col_src = (int*)alloc((size_t)Etot * sizeof(int));
  const int slots1 = (F_IN >> 5) * 128;     // per weight, K=F_IN
  const int slots2 = (64 >> 5) * 128;       // per weight, K=64
  _Float16* bp_l1  = (_Float16*)alloc((size_t)slots1 * 16 * sizeof(_Float16));
  _Float16* bp_r1  = (_Float16*)alloc((size_t)slots1 * 16 * sizeof(_Float16));
  _Float16* bp_l2  = (_Float16*)alloc((size_t)slots2 * 16 * sizeof(_Float16));
  _Float16* bp_r2  = (_Float16*)alloc((size_t)slots2 * 16 * sizeof(_Float16));
  (void)ws_size; (void)n_in; (void)out_size;

  // --- CSR build (once, reused by both layers) ---
  hipMemsetAsync(cursor, 0, (size_t)N * sizeof(int), stream);
  int eb = (Etot + 255) / 256;
  count_edges_kernel<<<eb, 256, 0, stream>>>(ei, E, N, cursor);
  scan_kernel<<<1, 1024, 0, stream>>>(cursor, row_ptr, cursor, N);
  scatter_edges_kernel<<<eb, 256, 0, stream>>>(ei, E, N, cursor, col_src);

  // --- Precision prep (runs after CSR build on the same stream) ---
  long long xpairs = (long long)N * F_IN / 2;
  f32_to_f16_kernel<<<(int)((xpairs + 255) / 256), 256, 0, stream>>>(x, xh, xpairs);
  pack_b_kernel<<<(slots1 + 255) / 256, 256, 0, stream>>>(w_l1, bp_l1, F_IN);
  pack_b_kernel<<<(slots1 + 255) / 256, 256, 0, stream>>>(w_r1, bp_r1, F_IN);
  pack_b_kernel<<<(slots2 + 255) / 256, 256, 0, stream>>>(w_l2, bp_l2, 64);
  pack_b_kernel<<<(slots2 + 255) / 256, 256, 0, stream>>>(w_r2, bp_r2, 64);

  // --- Layer 1 ---
  int gemm_waves  = (N + 15) / 16;
  int gemm_blocks = (gemm_waves + 3) / 4;   // 4 waves / 128-thread block
  gemm_wmma_f16<<<gemm_blocks, 128, 0, stream>>>(xh, bp_l1, xl, N, F_IN);
  gemm_wmma_f16<<<gemm_blocks, 128, 0, stream>>>(xh, bp_r1, xr, N, F_IN);
  int agg_blocks = (N + 3) / 4;             // 4 nodes (waves) / block
  gat_aggregate<<<agg_blocks, 128, 0, stream>>>(xl, xr, row_ptr, col_src,
                                                att1, b1, nullptr, h16, N, 1);

  // --- Layer 2 (reuse xl/xr buffers) ---
  gemm_wmma_f16<<<gemm_blocks, 128, 0, stream>>>(h16, bp_l2, xl, N, 64);
  gemm_wmma_f16<<<gemm_blocks, 128, 0, stream>>>(h16, bp_r2, xr, N, 64);
  gat_aggregate<<<agg_blocks, 128, 0, stream>>>(xl, xr, row_ptr, col_src,
                                                att2, b2, out, nullptr, N, 0);
}
